// Net_33947421508082
// MI455X (gfx1250) — matrix operand order repaired and checked by reading, not compiled, not run on hardware
//
#include <hip/hip_runtime.h>
#include <hip/hip_fp16.h>

typedef __attribute__((ext_vector_type(16))) _Float16 v16h;
typedef __attribute__((ext_vector_type(8)))  float    v8f;

#define F_IN  512
#define H_DIM 16
#define C_DIM 7
#define EPSN  1e-12f

// ---------- order-preserving float <-> uint for atomic segment-max ----------
__device__ __forceinline__ unsigned fkey(float f) {
    unsigned b = __float_as_uint(f);
    return (b & 0x80000000u) ? ~b : (b | 0x80000000u);
}
__device__ __forceinline__ float unfkey(unsigned u) {
    unsigned b = (u & 0x80000000u) ? (u & 0x7fffffffu) : ~u;
    return __uint_as_float(b);
}

// =====================================================================
// Kernel 1: h = relu(x @ W1^T + b1)   via v_wmma_f32_16x16x32_f16
// One wave (32 lanes) computes one 16x16 output tile (16 rows, 16 cols).
// =====================================================================
__global__ void lin1_relu_wmma(const float* __restrict__ x,
                               const float* __restrict__ W1,   // [16,512] row-major
                               const float* __restrict__ b1,   // [16]
                               float* __restrict__ h,          // [N,16]
                               int N)
{
    __shared__ _Float16 w1s[H_DIM * F_IN];   // 16 KB
    for (int i = threadIdx.x; i < H_DIM * F_IN; i += blockDim.x)
        w1s[i] = (_Float16)W1[i];
    __syncthreads();

    const int lane = threadIdx.x & 31;
    const int wv   = threadIdx.x >> 5;
    const int tile = blockIdx.x * (blockDim.x >> 5) + wv;   // wave-uniform
    if (tile * H_DIM >= N) return;                          // uniform exit

    const int ln   = lane & 15;     // A row index (M) and C column index (N)
    const int half = lane >> 4;

    int arow = tile * H_DIM + ln;
    if (arow >= N) arow = N - 1;    // clamped duplicate load; store is guarded
    const float*    __restrict__ xrow = x   + (size_t)arow * F_IN;
    const _Float16* __restrict__ wrow = w1s + ln * F_IN;    // B column n = ln

    v8f acc = {};
    #pragma unroll
    for (int kk = 0; kk < F_IN / 32; ++kk) {
        v16h a, b;
        const int kA = kk * 32 + half * 16;   // A: K = kA..kA+15 contiguous
        #pragma unroll
        for (int i = 0; i < 8; ++i) {
            a[i]     = (_Float16)xrow[kA + i];
            a[8 + i] = (_Float16)xrow[kA + 8 + i];
        }
        const int kB = kk * 32 + half * 16;   // B: K = kB..kB+15 contiguous
        #pragma unroll
        for (int i = 0; i < 16; ++i)
            b[i] = wrow[kB + i];
        acc = __builtin_amdgcn_wmma_f32_16x16x32_f16(
            /*neg_a=*/false, a, /*neg_b=*/false, b,
            /*c_mod=*/(short)0, acc, /*reuse_a=*/false, /*reuse_b=*/false);
    }

    const float bias = b1[ln];
    #pragma unroll
    for (int r = 0; r < 8; ++r) {
        const int M    = r + 8 * half;
        const int orow = tile * H_DIM + M;
        if (orow < N) {
            float v = acc[r] + bias;
            h[(size_t)orow * H_DIM + ln] = v > 0.0f ? v : 0.0f;
        }
    }
}

// =====================================================================
// Kernel 2 (per node): xn = h / max(|h|,eps); init hout=0, mkey=-inf, s=0
// =====================================================================
__global__ void norm_init(const float* __restrict__ hin,
                          float* __restrict__ xn,
                          float* __restrict__ hout,
                          unsigned* __restrict__ mkey,
                          float* __restrict__ ssum,
                          int N)
{
    int n = blockIdx.x * blockDim.x + threadIdx.x;
    if (n >= N) return;
    const float4* hp = (const float4*)(hin + (size_t)n * H_DIM);
    float4 v0 = hp[0], v1 = hp[1], v2 = hp[2], v3 = hp[3];
    float ss = v0.x*v0.x + v0.y*v0.y + v0.z*v0.z + v0.w*v0.w
             + v1.x*v1.x + v1.y*v1.y + v1.z*v1.z + v1.w*v1.w
             + v2.x*v2.x + v2.y*v2.y + v2.z*v2.z + v2.w*v2.w
             + v3.x*v3.x + v3.y*v3.y + v3.z*v3.z + v3.w*v3.w;
    float inv = 1.0f / fmaxf(sqrtf(ss), EPSN);
    float4* xp = (float4*)(xn + (size_t)n * H_DIM);
    xp[0] = make_float4(v0.x*inv, v0.y*inv, v0.z*inv, v0.w*inv);
    xp[1] = make_float4(v1.x*inv, v1.y*inv, v1.z*inv, v1.w*inv);
    xp[2] = make_float4(v2.x*inv, v2.y*inv, v2.z*inv, v2.w*inv);
    xp[3] = make_float4(v3.x*inv, v3.y*inv, v3.z*inv, v3.w*inv);
    float4 z = make_float4(0.f, 0.f, 0.f, 0.f);
    float4* op = (float4*)(hout + (size_t)n * H_DIM);
    op[0] = z; op[1] = z; op[2] = z; op[3] = z;
    mkey[n] = fkey(-3.0e38f);
    ssum[n] = 0.0f;
}

// =====================================================================
// Kernel 3 (4 lanes/edge): logit = beta * <xn[dst], xn[src]>;
// store logit, segment-max via atomicMax on monotone uint key.
// =====================================================================
__global__ void edge_logit(const float* __restrict__ xn,
                           const int* __restrict__ srcI,
                           const int* __restrict__ dstI,
                           const float* __restrict__ betaP,   // nullptr => 1.0
                           float* __restrict__ elog,
                           unsigned* __restrict__ mkey,
                           long long EN, long long E)
{
    long long t = (long long)blockIdx.x * blockDim.x + threadIdx.x;
    long long e = t >> 2;
    int q = (int)(t & 3);
    if (e >= EN) return;
    int s, d;
    if (e < E) { s = srcI[e]; d = dstI[e]; }
    else       { s = d = (int)(e - E); }          // self-loop
    float4 a = ((const float4*)(xn + (size_t)s * H_DIM))[q];
    float4 c = ((const float4*)(xn + (size_t)d * H_DIM))[q];
    float p = a.x*c.x + a.y*c.y + a.z*c.z + a.w*c.w;
    p += __shfl_xor(p, 1, 4);
    p += __shfl_xor(p, 2, 4);
    if (q == 0) {
        float beta = betaP ? betaP[0] : 1.0f;
        float lg = beta * p;
        elog[e] = lg;
        atomicMax(mkey + d, fkey(lg));
    }
}

// =====================================================================
// Kernel 4 (1 thread/edge): w = exp(logit - max[dst]); segment-sum
// =====================================================================
__global__ void edge_expsum(float* __restrict__ elog,
                            const unsigned* __restrict__ mkey,
                            const int* __restrict__ dstI,
                            float* __restrict__ ssum,
                            long long EN, long long E)
{
    long long e = (long long)blockIdx.x * blockDim.x + threadIdx.x;
    if (e >= EN) return;
    int d = (e < E) ? dstI[e] : (int)(e - E);
    float w = expf(elog[e] - unfkey(mkey[d]));
    elog[e] = w;
    atomicAdd(ssum + d, w);
}

// =====================================================================
// Kernel 5 (16 lanes/edge): out[dst] += (w/s[dst]) * h[src]
// =====================================================================
__global__ void edge_scatter(const float* __restrict__ elog,
                             const float* __restrict__ ssum,
                             const float* __restrict__ hin,
                             const int* __restrict__ srcI,
                             const int* __restrict__ dstI,
                             float* __restrict__ hout,
                             long long EN, long long E)
{
    long long t = (long long)blockIdx.x * blockDim.x + threadIdx.x;
    long long e = t >> 4;
    int f = (int)(t & 15);
    if (e >= EN) return;
    int s, d;
    if (e < E) { s = srcI[e]; d = dstI[e]; }
    else       { s = d = (int)(e - E); }
    float alpha = elog[e] / ssum[d];
    atomicAdd(hout + (size_t)d * H_DIM + f, alpha * hin[(size_t)s * H_DIM + f]);
}

// =====================================================================
// Kernel 6: logits = h @ W2^T + b2 ; log_softmax over 7 classes
// =====================================================================
__global__ void lin2_logsoftmax(const float* __restrict__ hin,
                                const float* __restrict__ W2,  // [7,16]
                                const float* __restrict__ b2,  // [7]
                                float* __restrict__ out,       // [N,7]
                                int N)
{
    int n = blockIdx.x * blockDim.x + threadIdx.x;
    if (n >= N) return;
    float hr[H_DIM];
    const float4* hp = (const float4*)(hin + (size_t)n * H_DIM);
    #pragma unroll
    for (int i = 0; i < 4; ++i) {
        float4 v = hp[i];
        hr[4*i+0] = v.x; hr[4*i+1] = v.y; hr[4*i+2] = v.z; hr[4*i+3] = v.w;
    }
    float lg[C_DIM];
    float mx = -3.0e38f;
    #pragma unroll
    for (int c = 0; c < C_DIM; ++c) {
        float acc = b2[c];
        #pragma unroll
        for (int k = 0; k < H_DIM; ++k)
            acc += hr[k] * W2[c * H_DIM + k];
        lg[c] = acc;
        mx = fmaxf(mx, acc);
    }
    float sum = 0.0f;
    #pragma unroll
    for (int c = 0; c < C_DIM; ++c)
        sum += expf(lg[c] - mx);
    float lse = mx + logf(sum);
    #pragma unroll
    for (int c = 0; c < C_DIM; ++c)
        out[(size_t)n * C_DIM + c] = lg[c] - lse;
}

// =====================================================================
extern "C" void kernel_launch(void* const* d_in, const int* in_sizes, int n_in,
                              void* d_out, int out_size, void* d_ws, size_t ws_size,
                              hipStream_t stream)
{
    const float* x     = (const float*)d_in[0];
    const int*   ei    = (const int*)  d_in[1];
    const float* W1    = (const float*)d_in[2];
    const float* b1    = (const float*)d_in[3];
    const float* beta2 = (const float*)d_in[4];
    const float* beta3 = (const float*)d_in[5];
    const float* beta4 = (const float*)d_in[6];
    const float* W2    = (const float*)d_in[7];
    const float* b2    = (const float*)d_in[8];

    const int       N  = in_sizes[0] / F_IN;
    const long long E  = in_sizes[1] / 2;
    const long long EN = E + N;
    const int* srcI = ei;
    const int* dstI = ei + E;

    // workspace carve-out (256B aligned slices)
    char* ws = (char*)d_ws;
    size_t off = 0;
    auto carve = [&](size_t bytes) -> void* {
        void* p = ws + off;
        off += (bytes + 255) & ~(size_t)255;
        return p;
    };
    float*    hA   = (float*)   carve((size_t)N * H_DIM * sizeof(float));
    float*    hB   = (float*)   carve((size_t)N * H_DIM * sizeof(float));
    float*    xn   = (float*)   carve((size_t)N * H_DIM * sizeof(float));
    unsigned* mkey = (unsigned*)carve((size_t)N * sizeof(unsigned));
    float*    ssum = (float*)   carve((size_t)N * sizeof(float));
    float*    elog = (float*)   carve((size_t)EN * sizeof(float));
    (void)ws_size; (void)n_in; (void)out_size;

    // ---- lin1 + relu (WMMA) ----
    const int tiles = (N + H_DIM - 1) / H_DIM;
    const int wavesPerBlock = 8;                         // 256 threads
    const int blk1 = (tiles + wavesPerBlock - 1) / wavesPerBlock;
    lin1_relu_wmma<<<blk1, 32 * wavesPerBlock, 0, stream>>>(x, W1, b1, hA, N);

    // ---- 4 AGNN propagation rounds ----
    const float* betas[4] = { nullptr, beta2, beta3, beta4 };
    float* hin = hA;
    float* hout = hB;
    const unsigned nblkN  = (unsigned)((N + 255) / 256);
    const unsigned nblkE1 = (unsigned)((EN + 255) / 256);
    const unsigned nblkE4 = (unsigned)((4 * EN + 255) / 256);
    const unsigned nblkE16= (unsigned)((16 * EN + 255) / 256);
    for (int r = 0; r < 4; ++r) {
        norm_init  <<<nblkN,   256, 0, stream>>>(hin, xn, hout, mkey, ssum, N);
        edge_logit <<<nblkE4,  256, 0, stream>>>(xn, srcI, dstI, betas[r], elog, mkey, EN, E);
        edge_expsum<<<nblkE1,  256, 0, stream>>>(elog, mkey, dstI, ssum, EN, E);
        edge_scatter<<<nblkE16,256, 0, stream>>>(elog, ssum, hin, srcI, dstI, hout, EN, E);
        float* tmp = hin; hin = hout; hout = tmp;
    }

    // ---- lin2 + log_softmax ----
    lin2_logsoftmax<<<nblkN, 256, 0, stream>>>(hin, W2, b2, (float*)d_out, N);
}
